// Transformer_block_34565896798699
// MI455X (gfx1250) — matrix operand order repaired and checked by reading, not compiled, not run on hardware
//
#include <hip/hip_runtime.h>

// ---------------------------------------------------------------------------
// Transformer-XL block for MI455X (gfx1250, wave32, WMMA + TDM).
// All GEMMs: bf16 x bf16 -> f32 via v_wmma_f32_16x16x32_bf16.
// Feed-forward GEMMs stage A/B tiles into LDS with tensor_load_to_lds (TDM),
// double-buffered, synced with s_wait_tensorcnt + workgroup barriers.
// ---------------------------------------------------------------------------

typedef __attribute__((ext_vector_type(16))) __bf16 v16bf;
typedef __attribute__((ext_vector_type(8)))  float  v8f;
typedef unsigned int u32x4 __attribute__((ext_vector_type(4)));
typedef int          i32x4v __attribute__((ext_vector_type(4)));
typedef int          i32x8v __attribute__((ext_vector_type(8)));

#define MODE_F32    0
#define MODE_BF16   1
#define MODE_BF16_T 2
#define MODE_QUV    3
#define MODE_SCORE  4

#define ACT_NONE 0
#define ACT_GELU 1
#define ACT_SIG  2

#if __has_builtin(__builtin_amdgcn_tensor_load_to_lds) && \
    __has_builtin(__builtin_amdgcn_s_wait_tensorcnt)
#define HAVE_TDM 1
#endif

__device__ __forceinline__ unsigned short f2bf(float f) {
  unsigned int u = __float_as_uint(f);
  u += 0x7FFFu + ((u >> 16) & 1u);            // round-to-nearest-even
  return (unsigned short)(u >> 16);
}
__device__ __forceinline__ float gelu_exact(float x) {
  return 0.5f * x * (1.0f + erff(x * 0.70710678118654752f));
}
__device__ __forceinline__ float sigmoid_f(float x) {
  return 1.0f / (1.0f + expf(-x));
}

union ABFrag {
  uint4 q[2];
  unsigned int d[8];
  v16bf v;
};

// Load a 16-element bf16 fragment from two 16-byte chunks.
__device__ __forceinline__ v16bf load_frag16(const unsigned short* p0,
                                             const unsigned short* p1) {
  ABFrag f;
  f.q[0] = *(const uint4*)p0;
  f.q[1] = *(const uint4*)p1;
  return f.v;
}

#ifdef HAVE_TDM
// Issue one TDM 2-D tile load (bf16 elements) into LDS at byte offset
// lds_byte_off. Descriptor per CDNA5 ISA §8.3/8.4: group0 = {count=1,
// lds_addr, global_addr, type=2}; group1 = {data_size=2B, dims/strides};
// groups 2/3 zero (2-D tensor, tile_dim2 = 0).
__device__ __forceinline__ void tdm_load_tile_2d(
    unsigned lds_byte_off, const unsigned short* gsrc,
    unsigned tile_d0, unsigned tile_d1,
    unsigned long long stride_d0_elems,
    unsigned tensor_d0, unsigned tensor_d1) {
  unsigned long long ga = (unsigned long long)(size_t)gsrc;
  u32x4 g0;
  g0[0] = 1u;                                            // count=1 (user D#)
  g0[1] = lds_byte_off;                                  // LDS dest (bytes)
  g0[2] = (unsigned)(ga & 0xFFFFFFFFu);                  // global_addr[31:0]
  g0[3] = (unsigned)((ga >> 32) & 0x01FFFFFFu)           // global_addr[56:32]
          | (2u << 30);                                  // type=2 ("image")
  i32x8v g1;
  g1[0] = (int)(1u << 16);                               // data_size=1 -> 2B
  g1[1] = (int)((tensor_d0 & 0xFFFFu) << 16);            // tensor_dim0[15:0]
  g1[2] = (int)((tensor_d0 >> 16) | ((tensor_d1 & 0xFFFFu) << 16));
  g1[3] = (int)((tensor_d1 >> 16) | ((tile_d0 & 0xFFFFu) << 16));
  g1[4] = (int)(tile_d1 & 0xFFFFu);                      // tile_dim1; dim2=0
  g1[5] = (int)(stride_d0_elems & 0xFFFFFFFFull);        // dim0_stride lo
  g1[6] = (int)((stride_d0_elems >> 32) & 0xFFFFull);    // dim0_stride hi
  g1[7] = 0;                                             // dim1_stride = 0
  i32x4v gz = {0, 0, 0, 0};
#if __clang_major__ >= 23
  i32x8v gz8 = {0, 0, 0, 0, 0, 0, 0, 0};
  __builtin_amdgcn_tensor_load_to_lds(g0, g1, gz, gz, gz8, 0);
#else
  __builtin_amdgcn_tensor_load_to_lds(g0, g1, gz, gz, 0);
#endif
}
#endif

// ---------------------------------------------------------------------------
// Generic NT GEMM (direct global loads): C[m,n] = sum_k A[m,k] * Bt[n,k].
// Block = 128 threads = 4 waves; wave computes 16(M) x 64(N); K step 32.
// ---------------------------------------------------------------------------
struct GemmArgs {
  const unsigned short* A;
  const unsigned short* Bt;
  float* outF;
  unsigned short* outB;
  unsigned short* outB2;
  const float* bias;
  const float* uvec;
  const float* vvec;
  const float* braw;     // MODE_SCORE: raw rel scores, S x N (f32)
  long long sAz, sBz, sOz;
  int lda, ldb, ldo;
  int M, N, K;
  int mode, act;
  int maskM;             // MODE_SCORE: mask if n > m + maskM
  int shift;             // MODE_SCORE: braw col = n + shift - m
  float scale;           // MODE_SCORE: 1/sqrt(D)
};

__global__ __launch_bounds__(128) void gemm_nt_bf16(GemmArgs g) {
  const int tid  = threadIdx.x;
  const int lane = tid & 31;
  const int wav  = tid >> 5;
  const int hf   = lane >> 4;
  const int l16  = lane & 15;
  const int m0 = blockIdx.y * 64 + wav * 16;
  const int n0 = blockIdx.x * 64;
  const long long z = blockIdx.z;

  const unsigned short* A  = g.A  + z * g.sAz;
  const unsigned short* Bt = g.Bt + z * g.sBz;

  const unsigned short* arow  = A  + (size_t)(m0 + l16) * g.lda;
  const unsigned short* brow0 = Bt + (size_t)(n0 +  0 + l16) * g.ldb;
  const unsigned short* brow1 = Bt + (size_t)(n0 + 16 + l16) * g.ldb;
  const unsigned short* brow2 = Bt + (size_t)(n0 + 32 + l16) * g.ldb;
  const unsigned short* brow3 = Bt + (size_t)(n0 + 48 + l16) * g.ldb;

  v8f acc0 = {}, acc1 = {}, acc2 = {}, acc3 = {};

  for (int k0 = 0; k0 < g.K; k0 += 32) {
    const unsigned short* ap = arow + k0 + hf * 16;
    const int kb = k0 + hf * 16;
    __builtin_prefetch(ap + 64, 0, 0);
    __builtin_prefetch(brow0 + kb + 64, 0, 0);
    v16bf av = load_frag16(ap, ap + 8);
    v16bf b0 = load_frag16(brow0 + kb, brow0 + kb + 8);
    v16bf b1 = load_frag16(brow1 + kb, brow1 + kb + 8);
    v16bf b2 = load_frag16(brow2 + kb, brow2 + kb + 8);
    v16bf b3 = load_frag16(brow3 + kb, brow3 + kb + 8);
    acc0 = __builtin_amdgcn_wmma_f32_16x16x32_bf16(false, av, false, b0,
                                                   (short)0, acc0, false, false);
    acc1 = __builtin_amdgcn_wmma_f32_16x16x32_bf16(false, av, false, b1,
                                                   (short)0, acc1, false, false);
    acc2 = __builtin_amdgcn_wmma_f32_16x16x32_bf16(false, av, false, b2,
                                                   (short)0, acc2, false, false);
    acc3 = __builtin_amdgcn_wmma_f32_16x16x32_bf16(false, av, false, b3,
                                                   (short)0, acc3, false, false);
  }

  const long long zo = z * g.sOz;
  v8f accs[4] = {acc0, acc1, acc2, acc3};
  for (int nt = 0; nt < 4; ++nt) {
    for (int r = 0; r < 8; ++r) {
      const int mm = m0 + hf * 8 + r;
      const int nn = n0 + nt * 16 + l16;
      float x = accs[nt][r];
      switch (g.mode) {
        case MODE_F32:
        case MODE_BF16:
        case MODE_BF16_T: {
          if (g.bias) x += g.bias[nn];
          if (g.act == ACT_GELU)     x = gelu_exact(x);
          else if (g.act == ACT_SIG) x = sigmoid_f(x);
          if (g.mode == MODE_F32)
            g.outF[zo + (size_t)mm * g.ldo + nn] = x;
          else if (g.mode == MODE_BF16)
            g.outB[zo + (size_t)mm * g.ldo + nn] = f2bf(x);
          else
            g.outB[zo + (size_t)nn * g.ldo + mm] = f2bf(x);
        } break;
        case MODE_QUV: {
          g.outB [zo + (size_t)mm * g.ldo + nn] = f2bf(x + g.uvec[nn]);
          g.outB2[zo + (size_t)mm * g.ldo + nn] = f2bf(x + g.vvec[nn]);
        } break;
        case MODE_SCORE: {
          const int j = nn + g.shift - mm;          // rel_enc_shift gather
          float bd = (j >= 0 && j < g.N) ? g.braw[(size_t)mm * g.N + j] : 0.0f;
          float sc = (x + bd) * g.scale;
          if (nn > mm + g.maskM) sc += -1.0e10f;    // causal+mem mask
          g.outF[(size_t)mm * g.ldo + nn] = sc;
        } break;
      }
    }
  }
}

// ---------------------------------------------------------------------------
// TDM-staged NT GEMM: 64x64 block tile; A(64x32) and B(64x32) bf16 tiles are
// DMA'd into LDS by the Tensor Data Mover, double-buffered. Wave w computes
// rows [w*16, w*16+16). Used for the big feed-forward GEMMs (z == 1,
// M/N/K multiples of 64).
// ---------------------------------------------------------------------------
__global__ __launch_bounds__(128) void gemm_nt_bf16_tdm(GemmArgs g) {
  // Single __shared__ object => group-segment offset 0, so TDM lds_addr is
  // just the byte offset into this array.  [buf][A=0/B=1][64 rows x 32 k]
  __shared__ unsigned short smem[2][2][64 * 32];

  const int tid  = threadIdx.x;
  const int lane = tid & 31;
  const int wav  = tid >> 5;
  const int hf   = lane >> 4;
  const int l16  = lane & 15;
  const int m0 = blockIdx.y * 64;
  const int n0 = blockIdx.x * 64;

  const unsigned short* Ablk = g.A  + (size_t)m0 * g.lda;
  const unsigned short* Bblk = g.Bt + (size_t)n0 * g.ldb;

#ifdef HAVE_TDM
  const unsigned tileBytes = 64u * 32u * 2u;         // 4 KB per tile
  if (tid == 0) {
    tdm_load_tile_2d(0 * tileBytes, Ablk, 32u, 64u,
                     (unsigned long long)g.lda, (unsigned)g.lda, 64u);
    tdm_load_tile_2d(1 * tileBytes, Bblk, 32u, 64u,
                     (unsigned long long)g.ldb, (unsigned)g.ldb, 64u);
  }
#else
  for (int i = tid; i < 64 * 4; i += 128) {          // 256 16-byte chunks
    const int r = i >> 2, c = (i & 3) * 8;
    *(uint4*)(&smem[0][0][0] + r * 32 + c) =
        *(const uint4*)(Ablk + (size_t)r * g.lda + c);
    *(uint4*)(&smem[0][1][0] + r * 32 + c) =
        *(const uint4*)(Bblk + (size_t)r * g.ldb + c);
  }
  __syncthreads();
#endif

  v8f acc0 = {}, acc1 = {}, acc2 = {}, acc3 = {};

  for (int k0 = 0, it = 0; k0 < g.K; k0 += 32, ++it) {
    const int cur = it & 1;
    const int nxt = cur ^ 1;

#ifdef HAVE_TDM
    if (tid == 0) __builtin_amdgcn_s_wait_tensorcnt(0);  // cur tiles landed
    __syncthreads();                                     // publish to all waves
    if (tid == 0 && (k0 + 32) < g.K) {                   // prefetch next tiles
      tdm_load_tile_2d((unsigned)((nxt * 2 + 0) * tileBytes), Ablk + k0 + 32,
                       32u, 64u, (unsigned long long)g.lda, (unsigned)g.lda,
                       64u);
      tdm_load_tile_2d((unsigned)((nxt * 2 + 1) * tileBytes), Bblk + k0 + 32,
                       32u, 64u, (unsigned long long)g.ldb, (unsigned)g.ldb,
                       64u);
    }
#endif

    const unsigned short* sA = &smem[cur][0][0];
    const unsigned short* sB = &smem[cur][1][0];
    const unsigned short* ap = sA + (wav * 16 + l16) * 32 + hf * 16;
    const int kb = hf * 16;
    v16bf av = load_frag16(ap, ap + 8);
    const unsigned short* b0p = sB + ( 0 + l16) * 32 + kb;
    const unsigned short* b1p = sB + (16 + l16) * 32 + kb;
    const unsigned short* b2p = sB + (32 + l16) * 32 + kb;
    const unsigned short* b3p = sB + (48 + l16) * 32 + kb;
    v16bf b0 = load_frag16(b0p, b0p + 8);
    v16bf b1 = load_frag16(b1p, b1p + 8);
    v16bf b2 = load_frag16(b2p, b2p + 8);
    v16bf b3 = load_frag16(b3p, b3p + 8);
    acc0 = __builtin_amdgcn_wmma_f32_16x16x32_bf16(false, av, false, b0,
                                                   (short)0, acc0, false, false);
    acc1 = __builtin_amdgcn_wmma_f32_16x16x32_bf16(false, av, false, b1,
                                                   (short)0, acc1, false, false);
    acc2 = __builtin_amdgcn_wmma_f32_16x16x32_bf16(false, av, false, b2,
                                                   (short)0, acc2, false, false);
    acc3 = __builtin_amdgcn_wmma_f32_16x16x32_bf16(false, av, false, b3,
                                                   (short)0, acc3, false, false);
    __syncthreads();                                  // reads done before reuse

#ifndef HAVE_TDM
    if ((k0 + 32) < g.K) {
      for (int i = tid; i < 64 * 4; i += 128) {
        const int r = i >> 2, c = (i & 3) * 8;
        *(uint4*)(&smem[nxt][0][0] + r * 32 + c) =
            *(const uint4*)(Ablk + (size_t)r * g.lda + k0 + 32 + c);
        *(uint4*)(&smem[nxt][1][0] + r * 32 + c) =
            *(const uint4*)(Bblk + (size_t)r * g.ldb + k0 + 32 + c);
      }
      __syncthreads();
    }
#endif
  }

  v8f accs[4] = {acc0, acc1, acc2, acc3};
  for (int nt = 0; nt < 4; ++nt) {
    for (int r = 0; r < 8; ++r) {
      const int mm = m0 + wav * 16 + hf * 8 + r;
      const int nn = n0 + nt * 16 + l16;
      float x = accs[nt][r];
      if (g.bias) x += g.bias[nn];
      if (g.act == ACT_GELU)     x = gelu_exact(x);
      else if (g.act == ACT_SIG) x = sigmoid_f(x);
      if (g.mode == MODE_BF16)
        g.outB[(size_t)mm * g.ldo + nn] = f2bf(x);
      else
        g.outF[(size_t)mm * g.ldo + nn] = x;
    }
  }
}

// ---------------------------------------------------------------------------
// attn(f32) x v(bf16, pre-transposed per head) -> ao(bf16). A converted to
// bf16 on the fly. z = b*H + h.
// ---------------------------------------------------------------------------
struct AttnGemmArgs {
  const float* attn;            // (B*H, S, F)
  const unsigned short* vt;     // (B, D, F): vt[b][n][f]
  unsigned short* ao;           // (B, S, D)
  int S, F, D, H;
};

__global__ __launch_bounds__(128) void gemm_attn_v(AttnGemmArgs g) {
  const int tid  = threadIdx.x;
  const int lane = tid & 31;
  const int wav  = tid >> 5;
  const int hf   = lane >> 4;
  const int l16  = lane & 15;
  const int z = blockIdx.z;
  const int b = z / g.H, h = z % g.H;
  const int m0 = blockIdx.y * 64 + wav * 16;
  const int n0 = blockIdx.x * 64;               // 0 (N = 64)

  const float* A = g.attn + (size_t)z * g.S * g.F;
  const unsigned short* Bt =
      g.vt + (size_t)b * g.D * g.F + (size_t)(h * 64) * g.F;
  unsigned short* out = g.ao + (size_t)b * g.S * g.D + h * 64;

  const float* arow = A + (size_t)(m0 + l16) * g.F;
  const unsigned short* brow0 = Bt + (size_t)(n0 +  0 + l16) * g.F;
  const unsigned short* brow1 = Bt + (size_t)(n0 + 16 + l16) * g.F;
  const unsigned short* brow2 = Bt + (size_t)(n0 + 32 + l16) * g.F;
  const unsigned short* brow3 = Bt + (size_t)(n0 + 48 + l16) * g.F;

  v8f acc0 = {}, acc1 = {}, acc2 = {}, acc3 = {};

  for (int k0 = 0; k0 < g.F; k0 += 32) {
    const float* ap = arow + k0 + hf * 16;
    float4 f0 = *(const float4*)(ap);
    float4 f1 = *(const float4*)(ap + 4);
    float4 f2 = *(const float4*)(ap + 8);
    float4 f3 = *(const float4*)(ap + 12);
    ABFrag a;
    a.d[0] = (unsigned)f2bf(f0.x) | ((unsigned)f2bf(f0.y) << 16);
    a.d[1] = (unsigned)f2bf(f0.z) | ((unsigned)f2bf(f0.w) << 16);
    a.d[2] = (unsigned)f2bf(f1.x) | ((unsigned)f2bf(f1.y) << 16);
    a.d[3] = (unsigned)f2bf(f1.z) | ((unsigned)f2bf(f1.w) << 16);
    a.d[4] = (unsigned)f2bf(f2.x) | ((unsigned)f2bf(f2.y) << 16);
    a.d[5] = (unsigned)f2bf(f2.z) | ((unsigned)f2bf(f2.w) << 16);
    a.d[6] = (unsigned)f2bf(f3.x) | ((unsigned)f2bf(f3.y) << 16);
    a.d[7] = (unsigned)f2bf(f3.z) | ((unsigned)f2bf(f3.w) << 16);
    const int kb = k0 + hf * 16;
    v16bf b0 = load_frag16(brow0 + kb, brow0 + kb + 8);
    v16bf b1 = load_frag16(brow1 + kb, brow1 + kb + 8);
    v16bf b2 = load_frag16(brow2 + kb, brow2 + kb + 8);
    v16bf b3 = load_frag16(brow3 + kb, brow3 + kb + 8);
    acc0 = __builtin_amdgcn_wmma_f32_16x16x32_bf16(false, a.v, false, b0,
                                                   (short)0, acc0, false, false);
    acc1 = __builtin_amdgcn_wmma_f32_16x16x32_bf16(false, a.v, false, b1,
                                                   (short)0, acc1, false, false);
    acc2 = __builtin_amdgcn_wmma_f32_16x16x32_bf16(false, a.v, false, b2,
                                                   (short)0, acc2, false, false);
    acc3 = __builtin_amdgcn_wmma_f32_16x16x32_bf16(false, a.v, false, b3,
                                                   (short)0, acc3, false, false);
  }

  v8f accs[4] = {acc0, acc1, acc2, acc3};
  for (int nt = 0; nt < 4; ++nt)
    for (int r = 0; r < 8; ++r) {
      const int mm = m0 + hf * 8 + r;
      const int nn = n0 + nt * 16 + l16;
      out[(size_t)mm * g.D + nn] = f2bf(accs[nt][r]);
    }
}

// ---------------------------------------------------------------------------
// LayerNorm over rows of concat(mem, inputs), bf16 output.
// ---------------------------------------------------------------------------
__global__ __launch_bounds__(256) void ln_concat_kernel(
    const float* inputs, const float* mem, const float* gw, const float* bw,
    unsigned short* out, int S, int Mlen, int D) {
  const int row = blockIdx.x;          // B*F rows
  const int F = S + Mlen;
  const int b = row / F, f = row % F;
  const float* src = (f < Mlen)
      ? (mem + ((size_t)b * Mlen + f) * D)
      : (inputs + ((size_t)b * S + (f - Mlen)) * D);
  __shared__ float red[256];
  const int tid = threadIdx.x;
  float s = 0.f;
  for (int i = tid; i < D; i += 256) s += src[i];
  red[tid] = s; __syncthreads();
  for (int st = 128; st > 0; st >>= 1) {
    if (tid < st) red[tid] += red[tid + st];
    __syncthreads();
  }
  const float mu = red[0] / (float)D;
  __syncthreads();
  float q = 0.f;
  for (int i = tid; i < D; i += 256) { float d = src[i] - mu; q += d * d; }
  red[tid] = q; __syncthreads();
  for (int st = 128; st > 0; st >>= 1) {
    if (tid < st) red[tid] += red[tid + st];
    __syncthreads();
  }
  const float rstd = rsqrtf(red[0] / (float)D + 1e-5f);
  unsigned short* o = out + (size_t)row * D;
  for (int i = tid; i < D; i += 256)
    o[i] = f2bf((src[i] - mu) * rstd * gw[i] + bw[i]);
}

__global__ __launch_bounds__(256) void ln_rows_kernel(
    const float* src, const float* gw, const float* bw,
    unsigned short* out, int D) {
  const size_t row = blockIdx.x;
  const float* p = src + row * D;
  __shared__ float red[256];
  const int tid = threadIdx.x;
  float s = 0.f;
  for (int i = tid; i < D; i += 256) s += p[i];
  red[tid] = s; __syncthreads();
  for (int st = 128; st > 0; st >>= 1) {
    if (tid < st) red[tid] += red[tid + st];
    __syncthreads();
  }
  const float mu = red[0] / (float)D;
  __syncthreads();
  float q = 0.f;
  for (int i = tid; i < D; i += 256) { float d = p[i] - mu; q += d * d; }
  red[tid] = q; __syncthreads();
  for (int st = 128; st > 0; st >>= 1) {
    if (tid < st) red[tid] += red[tid + st];
    __syncthreads();
  }
  const float rstd = rsqrtf(red[0] / (float)D + 1e-5f);
  unsigned short* o = out + row * D;
  for (int i = tid; i < D; i += 256)
    o[i] = f2bf((p[i] - mu) * rstd * gw[i] + bw[i]);
}

// ---------------------------------------------------------------------------
// In-place row softmax over attn; rowmax[row] = max of softmax row = 1/sumexp.
// ---------------------------------------------------------------------------
__global__ __launch_bounds__(256) void softmax_kernel(float* attn,
                                                      float* rowmax, int F) {
  const size_t row = blockIdx.x;
  float* p = attn + row * (size_t)F;
  __shared__ float red[256];
  const int tid = threadIdx.x;
  const int nper = F >> 8;                 // 8 for F=2048
  float v[8];
  float mx = -3.4e38f;
  for (int i = 0; i < nper; ++i) { v[i] = p[tid + i * 256]; mx = fmaxf(mx, v[i]); }
  red[tid] = mx; __syncthreads();
  for (int st = 128; st > 0; st >>= 1) {
    if (tid < st) red[tid] = fmaxf(red[tid], red[tid + st]);
    __syncthreads();
  }
  const float rmax = red[0];
  __syncthreads();
  float s = 0.f;
  for (int i = 0; i < nper; ++i) { v[i] = expf(v[i] - rmax); s += v[i]; }
  red[tid] = s; __syncthreads();
  for (int st = 128; st > 0; st >>= 1) {
    if (tid < st) red[tid] += red[tid + st];
    __syncthreads();
  }
  const float inv = 1.0f / red[0];
  for (int i = 0; i < nper; ++i) p[tid + i * 256] = v[i] * inv;
  if (tid == 0) rowmax[row] = inv;         // exp(0) * inv
}

__global__ __launch_bounds__(256) void loss_kernel(const float* rowmax,
                                                   float* out, int BH, int S) {
  __shared__ float red[256];
  __shared__ float accum;
  const int tid = threadIdx.x;
  if (tid == 0) accum = 0.f;
  __syncthreads();
  for (int bh = 0; bh < BH; ++bh) {
    float m = -3.4e38f;
    for (int i = tid; i < S; i += 256)
      m = fmaxf(m, rowmax[(size_t)bh * S + i]);
    red[tid] = m; __syncthreads();
    for (int st = 128; st > 0; st >>= 1) {
      if (tid < st) red[tid] = fmaxf(red[tid], red[tid + st]);
      __syncthreads();
    }
    if (tid == 0) accum += red[0];
    __syncthreads();
  }
  if (tid == 0) *out = accum / (float)BH;
}

// ---------------------------------------------------------------------------
// Elementwise helpers.
// ---------------------------------------------------------------------------
__global__ void cvt_kernel(const float* s, unsigned short* d, size_t n) {
  size_t i = (size_t)blockIdx.x * 256 + threadIdx.x;
  if (i < n) d[i] = f2bf(s[i]);
}
__global__ void transpose_cvt_kernel(const float* W, unsigned short* Wt,
                                     int K, int N) {
  int n = blockIdx.x * 256 + threadIdx.x;
  int k = blockIdx.y;
  if (n < N) Wt[(size_t)n * K + k] = f2bf(W[(size_t)k * N + n]);
}
__global__ void g1_kernel(const float* inp, const float* sig1,
                          const float* rmha, float* g1, unsigned short* g1b,
                          size_t n) {
  size_t i = (size_t)blockIdx.x * 256 + threadIdx.x;
  if (i < n) {
    float v = inp[i] + sig1[i] * rmha[i];
    g1[i] = v;
    g1b[i] = f2bf(v);
  }
}
__global__ void out_kernel(const float* g1, const float* sig2,
                           const float* h2, float* out, size_t n) {
  size_t i = (size_t)blockIdx.x * 256 + threadIdx.x;
  if (i < n) out[i] = g1[i] + sig2[i] * h2[i];
}

// ---------------------------------------------------------------------------
// Host orchestration.
// ---------------------------------------------------------------------------
extern "C" void kernel_launch(void* const* d_in, const int* in_sizes, int n_in,
                              void* d_out, int out_size, void* d_ws,
                              size_t ws_size, hipStream_t stream) {
  (void)in_sizes; (void)n_in; (void)out_size; (void)ws_size;
  const int Bn = 2, S = 1024, Mlen = 1024, D = 1024, H = 16;
  const int F = S + Mlen;                          // 2048
  const size_t BSD = (size_t)Bn * S * D;
  const size_t BFD = (size_t)Bn * F * D;

  const float* inputs  = (const float*)d_in[0];
  const float* mem     = (const float*)d_in[1];
  const float* rel_enc = (const float*)d_in[2];
  /* d_in[3] = mask: recomputed analytically (f > s + Mlen) */
  const float* ln1_g = (const float*)d_in[4];
  const float* ln1_b = (const float*)d_in[5];
  const float* wq  = (const float*)d_in[6];
  const float* wke = (const float*)d_in[7];
  const float* wkr = (const float*)d_in[8];
  const float* wv  = (const float*)d_in[9];
  const float* wf  = (const float*)d_in[10];
  const float* u_p = (const float*)d_in[11];
  const float* v_p = (const float*)d_in[12];
  const float* ln2_g = (const float*)d_in[13];
  const float* ln2_b = (const float*)d_in[14];
  const float* w1  = (const float*)d_in[15];
  const float* b1  = (const float*)d_in[16];
  const float* w2  = (const float*)d_in[17];
  const float* b2  = (const float*)d_in[18];
  const float* wg1 = (const float*)d_in[19];
  const float* bg1 = (const float*)d_in[20];
  const float* wg2 = (const float*)d_in[21];
  const float* bg2 = (const float*)d_in[22];

  float* out_x = (float*)d_out;
  float* attn  = out_x + BSD;
  float* loss  = attn + (size_t)Bn * H * S * F;

  char* wsp = (char*)d_ws;
  size_t off = 0;
  auto alloc = [&](size_t bytes) -> void* {
    void* p = wsp + off;
    off += (bytes + 255) & ~(size_t)255;
    return p;
  };
  unsigned short* xt   = (unsigned short*)alloc(BFD * 2);
  unsigned short* inb  = (unsigned short*)alloc(BSD * 2);
  unsigned short* reb  = (unsigned short*)alloc((size_t)F * D * 2);
  unsigned short* wqT  = (unsigned short*)alloc((size_t)D * D * 2);
  unsigned short* wkeT = (unsigned short*)alloc((size_t)D * D * 2);
  unsigned short* wkrT = (unsigned short*)alloc((size_t)D * D * 2);
  unsigned short* wvT  = (unsigned short*)alloc((size_t)D * D * 2);
  unsigned short* wfT  = (unsigned short*)alloc((size_t)D * D * 2);
  unsigned short* w1T  = (unsigned short*)alloc((size_t)D * D * 2);
  unsigned short* w2T  = (unsigned short*)alloc((size_t)D * D * 2);
  unsigned short* wg1T = (unsigned short*)alloc((size_t)D * D * 2);
  unsigned short* wg2T = (unsigned short*)alloc((size_t)D * D * 2);
  unsigned short* qu   = (unsigned short*)alloc(BSD * 2);
  unsigned short* qv   = (unsigned short*)alloc(BSD * 2);
  unsigned short* kb   = (unsigned short*)alloc(BFD * 2);
  unsigned short* vt   = (unsigned short*)alloc((size_t)Bn * D * F * 2);
  unsigned short* Qr   = (unsigned short*)alloc((size_t)F * D * 2);
  float* braw   = (float*)alloc((size_t)S * F * 4);
  float* rowmax = (float*)alloc((size_t)Bn * H * S * 4);
  unsigned short* aob  = (unsigned short*)alloc(BSD * 2);
  float* rmha = (float*)alloc(BSD * 4);
  float* sig1 = (float*)alloc(BSD * 4);
  float* g1   = (float*)alloc(BSD * 4);
  unsigned short* g1b = (unsigned short*)alloc(BSD * 2);
  unsigned short* hln = (unsigned short*)alloc(BSD * 2);
  unsigned short* h1b = (unsigned short*)alloc(BSD * 2);
  float* h2   = (float*)alloc(BSD * 4);
  float* sig2 = (float*)alloc(BSD * 4);

  // ---- conversions ----
  cvt_kernel<<<dim3((unsigned)((BSD + 255) / 256)), 256, 0, stream>>>(
      inputs, inb, BSD);
  cvt_kernel<<<dim3((unsigned)(((size_t)F * D + 255) / 256)), 256, 0, stream>>>(
      rel_enc, reb, (size_t)F * D);
  {
    dim3 tg(D / 256, D);
    transpose_cvt_kernel<<<tg, 256, 0, stream>>>(wq,  wqT,  D, D);
    transpose_cvt_kernel<<<tg, 256, 0, stream>>>(wke, wkeT, D, D);
    transpose_cvt_kernel<<<tg, 256, 0, stream>>>(wkr, wkrT, D, D);
    transpose_cvt_kernel<<<tg, 256, 0, stream>>>(wv,  wvT,  D, D);
    transpose_cvt_kernel<<<tg, 256, 0, stream>>>(wf,  wfT,  D, D);
    transpose_cvt_kernel<<<tg, 256, 0, stream>>>(w1,  w1T,  D, D);
    transpose_cvt_kernel<<<tg, 256, 0, stream>>>(w2,  w2T,  D, D);
    transpose_cvt_kernel<<<tg, 256, 0, stream>>>(wg1, wg1T, D, D);
    transpose_cvt_kernel<<<tg, 256, 0, stream>>>(wg2, wg2T, D, D);
  }

  // ---- x_tilde = LN(concat(mem, inputs)) -> bf16 ----
  ln_concat_kernel<<<Bn * F, 256, 0, stream>>>(inputs, mem, ln1_g, ln1_b, xt,
                                               S, Mlen, D);

  // ---- q = x @ wq (+u / +v epilogue) ----
  {
    GemmArgs ga{};
    ga.A = xt + (size_t)Mlen * D; ga.lda = D; ga.sAz = (long long)F * D;
    ga.Bt = wqT; ga.ldb = D; ga.sBz = 0;
    ga.M = S; ga.N = D; ga.K = D;
    ga.mode = MODE_QUV; ga.act = ACT_NONE;
    ga.uvec = u_p; ga.vvec = v_p;
    ga.outB = qu; ga.outB2 = qv; ga.ldo = D; ga.sOz = (long long)S * D;
    gemm_nt_bf16<<<dim3(D / 64, S / 64, Bn), 128, 0, stream>>>(ga);
  }
  // ---- k = x_tilde @ wke ----
  {
    GemmArgs ga{};
    ga.A = xt; ga.lda = D; ga.sAz = (long long)F * D;
    ga.Bt = wkeT; ga.ldb = D;
    ga.M = F; ga.N = D; ga.K = D;
    ga.mode = MODE_BF16; ga.outB = kb; ga.ldo = D; ga.sOz = (long long)F * D;
    gemm_nt_bf16<<<dim3(D / 64, F / 64, Bn), 128, 0, stream>>>(ga);
  }
  // ---- v = x_tilde @ wv, stored transposed per (b): vt[b][n][f] ----
  {
    GemmArgs ga{};
    ga.A = xt; ga.lda = D; ga.sAz = (long long)F * D;
    ga.Bt = wvT; ga.ldb = D;
    ga.M = F; ga.N = D; ga.K = D;
    ga.mode = MODE_BF16_T; ga.outB = vt; ga.ldo = F; ga.sOz = (long long)D * F;
    gemm_nt_bf16<<<dim3(D / 64, F / 64, Bn), 128, 0, stream>>>(ga);
  }
  // ---- Qr = rel_enc @ wkr ----
  {
    GemmArgs ga{};
    ga.A = reb; ga.lda = D;
    ga.Bt = wkrT; ga.ldb = D;
    ga.M = F; ga.N = D; ga.K = D;
    ga.mode = MODE_BF16; ga.outB = Qr; ga.ldo = D;
    gemm_nt_bf16<<<dim3(D / 64, F / 64, 1), 128, 0, stream>>>(ga);
  }

  // ---- per-(b,h): Braw then score (A_C + shifted B_D, scaled, masked) ----
  for (int bh = 0; bh < Bn * H; ++bh) {
    const int b = bh / H, h = bh % H;
    {
      GemmArgs ga{};
      ga.A = qv + ((size_t)b * S * D + h * 64); ga.lda = D;
      ga.Bt = Qr + h * 64; ga.ldb = D;
      ga.M = S; ga.N = F; ga.K = 64;
      ga.mode = MODE_F32; ga.outF = braw; ga.ldo = F;
      gemm_nt_bf16<<<dim3(F / 64, S / 64, 1), 128, 0, stream>>>(ga);
    }
    {
      GemmArgs ga{};
      ga.A = qu + ((size_t)b * S * D + h * 64); ga.lda = D;
      ga.Bt = kb + ((size_t)b * F * D + h * 64); ga.ldb = D;
      ga.M = S; ga.N = F; ga.K = 64;
      ga.mode = MODE_SCORE;
      ga.braw = braw;
      ga.outF = attn + (size_t)bh * S * F; ga.ldo = F;
      ga.scale = 0.03125f;        // 1/sqrt(1024)
      ga.maskM = Mlen;
      ga.shift = S - 1;
      gemm_nt_bf16<<<dim3(F / 64, S / 64, 1), 128, 0, stream>>>(ga);
    }
  }

  // ---- softmax (in place in d_out) + per-row max, then the loss ----
  softmax_kernel<<<Bn * H * S, 256, 0, stream>>>(attn, rowmax, F);
  loss_kernel<<<1, 256, 0, stream>>>(rowmax, loss, Bn * H, S);

  // ---- ao = attn @ v ----
  {
    AttnGemmArgs aa{attn, vt, aob, S, F, D, H};
    gemm_attn_v<<<dim3(1, S / 64, Bn * H), 128, 0, stream>>>(aa);
  }
  // ---- rmha = gelu(ao @ wf)  (TDM-staged GEMM) ----
  {
    GemmArgs ga{};
    ga.A = aob; ga.lda = D;
    ga.Bt = wfT; ga.ldb = D;
    ga.M = Bn * S; ga.N = D; ga.K = D;
    ga.mode = MODE_F32; ga.act = ACT_GELU;
    ga.outF = rmha; ga.ldo = D;
    gemm_nt_bf16_tdm<<<dim3(D / 64, (Bn * S) / 64, 1), 128, 0, stream>>>(ga);
  }
  // ---- sig1 = sigmoid(inputs @ wg1 + bg1) ----
  {
    GemmArgs ga{};
    ga.A = inb; ga.lda = D;
    ga.Bt = wg1T; ga.ldb = D;
    ga.M = Bn * S; ga.N = D; ga.K = D;
    ga.mode = MODE_F32; ga.act = ACT_SIG; ga.bias = bg1;
    ga.outF = sig1; ga.ldo = D;
    gemm_nt_bf16_tdm<<<dim3(D / 64, (Bn * S) / 64, 1), 128, 0, stream>>>(ga);
  }
  // ---- g1 = inputs + sig1 * rmha ----
  g1_kernel<<<dim3((unsigned)((BSD + 255) / 256)), 256, 0, stream>>>(
      inputs, sig1, rmha, g1, g1b, BSD);
  // ---- h = LN2(g1) ----
  ln_rows_kernel<<<Bn * S, 256, 0, stream>>>(g1, ln2_g, ln2_b, hln, D);
  // ---- h1 = gelu(h @ w1 + b1) ----
  {
    GemmArgs ga{};
    ga.A = hln; ga.lda = D;
    ga.Bt = w1T; ga.ldb = D;
    ga.M = Bn * S; ga.N = D; ga.K = D;
    ga.mode = MODE_BF16; ga.act = ACT_GELU; ga.bias = b1;
    ga.outB = h1b; ga.ldo = D;
    gemm_nt_bf16_tdm<<<dim3(D / 64, (Bn * S) / 64, 1), 128, 0, stream>>>(ga);
  }
  // ---- h2 = gelu(h1 @ w2 + b2) ----
  {
    GemmArgs ga{};
    ga.A = h1b; ga.lda = D;
    ga.Bt = w2T; ga.ldb = D;
    ga.M = Bn * S; ga.N = D; ga.K = D;
    ga.mode = MODE_F32; ga.act = ACT_GELU; ga.bias = b2;
    ga.outF = h2; ga.ldo = D;
    gemm_nt_bf16_tdm<<<dim3(D / 64, (Bn * S) / 64, 1), 128, 0, stream>>>(ga);
  }
  // ---- sig2 = sigmoid(g1 @ wg2 + bg2) ----
  {
    GemmArgs ga{};
    ga.A = g1b; ga.lda = D;
    ga.Bt = wg2T; ga.ldb = D;
    ga.M = Bn * S; ga.N = D; ga.K = D;
    ga.mode = MODE_F32; ga.act = ACT_SIG; ga.bias = bg2;
    ga.outF = sig2; ga.ldo = D;
    gemm_nt_bf16_tdm<<<dim3(D / 64, (Bn * S) / 64, 1), 128, 0, stream>>>(ga);
  }
  // ---- out = g1 + sig2 * h2 ----
  out_kernel<<<dim3((unsigned)((BSD + 255) / 256)), 256, 0, stream>>>(
      g1, sig2, h2, out_x, BSD);
}
